// EvoSNN_73693048865431
// MI455X (gfx1250) — matrix operand order repaired and checked by reading, not compiled, not run on hardware
//
#include <hip/hip_runtime.h>

typedef __attribute__((ext_vector_type(16))) _Float16 v16h;
typedef __attribute__((ext_vector_type(8)))  _Float16 v8h;
typedef __attribute__((ext_vector_type(8)))  float    v8f;

#define SNN_THR  1.0f
#define SNN_BETA 0.9f

__global__ __launch_bounds__(256, 1)
void snn_wmma_kernel(const float* __restrict__ x,
                     const float* __restrict__ W1,
                     const float* __restrict__ b1,
                     const float* __restrict__ W2,
                     const float* __restrict__ b2,
                     const float* __restrict__ W3,
                     const float* __restrict__ b3,
                     const int*   __restrict__ nsteps,
                     float* __restrict__ out,
                     int B)
{
    // Per-wave private LDS slices (8 waves / block): s2 transpose bounce + output permute.
    __shared__ __align__(16) _Float16 lds_s2[8][16 * 32];
    __shared__ __align__(16) float    lds_out[8][64];

    const int lane = threadIdx.x & 31;
    const int wib  = threadIdx.x >> 5;
    const int gw   = blockIdx.x * 8 + wib;
    const int base = gw * 16;                 // 16 batch rows per wave

    const int T      = nsteps[0];
    const int memOff = T * B * 2;             // mem_rec follows spk_rec

    const int n0 = lane & 15;                 // B/C-layout column within 16-wide tile
    const int r  = lane & 15;                 // A-layout row
    const int kq = (lane < 16) ? 0 : 8;       // A-layout K-quarter base (16-bit A, 16x32)
    const int kh = (lane < 16) ? 0 : 16;      // K-half base (unused; B fragments use kq order like A)
    const int mb = (lane < 16) ? 0 : 8;       // C-layout M base

    // ---- W2^T (64x32) as f16 B fragments: [ktile][ntile], each 32x16 ----
    v16h bw2[2][2];
#pragma unroll
    for (int nt = 0; nt < 2; ++nt)
#pragma unroll
      for (int kt = 0; kt < 2; ++kt)
#pragma unroll
        for (int e = 0; e < 16; ++e)
          bw2[kt][nt][e] = (_Float16)W2[(nt * 16 + n0) * 64 + kt * 32 + kq + e + (e & 8)];

    // ---- W3^T (32x2) as one 32x16 B fragment, columns >= 2 zeroed ----
    v16h bw3;
#pragma unroll
    for (int e = 0; e < 16; ++e)
      bw3[e] = (n0 < 2) ? (_Float16)W3[n0 * 32 + kq + e + (e & 8)] : (_Float16)0.0f;

    // ---- biases broadcast into C layout (value depends only on N column) ----
    v8f biasC2a, biasC2b, biasC3;
    {
      const float v0 = b2[n0];
      const float v1 = b2[16 + n0];
      const float v2 = (n0 < 2) ? b3[n0] : 0.0f;
#pragma unroll
      for (int g = 0; g < 8; ++g) { biasC2a[g] = v0; biasC2b[g] = v1; biasC3[g] = v2; }
    }

    // ---- cur1 = x @ W1^T + b1, stored in the 16-bit A-matrix layout (f32) ----
    float xr[11];
#pragma unroll
    for (int j = 0; j < 11; ++j) xr[j] = x[(base + r) * 11 + j];

    float cur1v[32], m1v[32];
#pragma unroll
    for (int i = 0; i < 32; ++i) {
      const int e = i & 15;
      const int k = (i >> 4) * 32 + kq + e + (e & 8);   // neuron index this slot holds
      float acc = b1[k];
#pragma unroll
      for (int j = 0; j < 11; ++j) acc += xr[j] * W1[k * 11 + j];
      cur1v[i] = acc;
      m1v[i]   = 0.0f;
    }

    float m2a[8], m2b[8], m3v[8];
#pragma unroll
    for (int g = 0; g < 8; ++g) { m2a[g] = 0.0f; m2b[g] = 0.0f; m3v[g] = 0.0f; }

    _Float16* myS2 = &lds_s2[wib][0];
    float*    myO  = &lds_out[wib][0];

#pragma unroll 1
    for (int t = 0; t < T; ++t) {
      // ---- layer 1: leaky + fire -> f16 A fragments (pure per-lane) ----
      v16h aLo, aHi;
#pragma unroll
      for (int e = 0; e < 16; ++e) {
        float m  = m1v[e];
        float rs = (m > SNN_THR) ? SNN_THR : 0.0f;
        m = SNN_BETA * m + cur1v[e] - rs;
        m1v[e] = m;
        aLo[e] = (m > SNN_THR) ? (_Float16)1.0f : (_Float16)0.0f;

        float mh = m1v[16 + e];
        float rh = (mh > SNN_THR) ? SNN_THR : 0.0f;
        mh = SNN_BETA * mh + cur1v[16 + e] - rh;
        m1v[16 + e] = mh;
        aHi[e] = (mh > SNN_THR) ? (_Float16)1.0f : (_Float16)0.0f;
      }

      // ---- layer 2 GEMM: cur2 = s1 @ W2^T + b2 (K=64 chained, N=32 in 2 tiles) ----
      v8f c2a = __builtin_amdgcn_wmma_f32_16x16x32_f16(false, aLo, false, bw2[0][0], (short)0, biasC2a, false, false);
      c2a     = __builtin_amdgcn_wmma_f32_16x16x32_f16(false, aHi, false, bw2[1][0], (short)0, c2a,     false, false);
      v8f c2b = __builtin_amdgcn_wmma_f32_16x16x32_f16(false, aLo, false, bw2[0][1], (short)0, biasC2b, false, false);
      c2b     = __builtin_amdgcn_wmma_f32_16x16x32_f16(false, aHi, false, bw2[1][1], (short)0, c2b,     false, false);

      // ---- layer 2: leaky + fire; stage s2 row-major (16x32 f16) in LDS ----
#pragma unroll
      for (int g = 0; g < 8; ++g) {
        const int m = mb + g;
        float ma = m2a[g];
        float ra = (ma > SNN_THR) ? SNN_THR : 0.0f;
        ma = SNN_BETA * ma + c2a[g] - ra;
        m2a[g] = ma;
        myS2[m * 32 + n0] = (ma > SNN_THR) ? (_Float16)1.0f : (_Float16)0.0f;

        float mv = m2b[g];
        float rb = (mv > SNN_THR) ? SNN_THR : 0.0f;
        mv = SNN_BETA * mv + c2b[g] - rb;
        m2b[g] = mv;
        myS2[m * 32 + 16 + n0] = (mv > SNN_THR) ? (_Float16)1.0f : (_Float16)0.0f;
      }
      __syncthreads();

      // ---- gather s2 back in A layout (two aligned 16B LDS reads per lane) ----
      v8h lo8 = *(const v8h*)&myS2[r * 32 + kq];
      v8h hi8 = *(const v8h*)&myS2[r * 32 + kq + 16];
      v16h a2 = __builtin_shufflevector(lo8, hi8, 0,1,2,3,4,5,6,7,8,9,10,11,12,13,14,15);

      // ---- layer 3 GEMM (N padded to 16; only cols 0..1 valid) ----
      v8f c3 = __builtin_amdgcn_wmma_f32_16x16x32_f16(false, a2, false, bw3, (short)0, biasC3, false, false);

      // ---- layer 3: leaky + fire; stage (s3, m3) for coalesced store ----
      if (n0 < 2) {
#pragma unroll
        for (int g = 0; g < 8; ++g) {
          float m  = m3v[g];
          float rs = (m > SNN_THR) ? SNN_THR : 0.0f;
          m = SNN_BETA * m + c3[g] - rs;
          m3v[g] = m;
          const int row = mb + g;
          myO[row * 2 + n0]      = (m > SNN_THR) ? 1.0f : 0.0f;  // spike
          myO[32 + row * 2 + n0] = m;                            // membrane
        }
      }
      __syncthreads();

      // ---- streaming (nontemporal) coalesced stores: 128B per wave per array ----
      const int o = t * (B * 2) + base * 2 + lane;
      __builtin_nontemporal_store(myO[lane],      out + o);
      __builtin_nontemporal_store(myO[32 + lane], out + memOff + o);
      __syncthreads();
    }
}

extern "C" void kernel_launch(void* const* d_in, const int* in_sizes, int n_in,
                              void* d_out, int out_size, void* d_ws, size_t ws_size,
                              hipStream_t stream) {
    const float* x   = (const float*)d_in[0];
    const float* W1  = (const float*)d_in[1];
    const float* b1  = (const float*)d_in[2];
    const float* W2  = (const float*)d_in[3];
    const float* b2  = (const float*)d_in[4];
    const float* W3  = (const float*)d_in[5];
    const float* b3  = (const float*)d_in[6];
    const int*   ns  = (const int*)d_in[7];
    float*       out = (float*)d_out;

    const int B = in_sizes[0] / 11;          // batch rows
    const int blocks = B / 128;              // 8 waves/block * 16 rows/wave
    snn_wmma_kernel<<<blocks, 256, 0, stream>>>(x, W1, b1, W2, b2, W3, b3, ns, out, B);
}
